// SchNetWrapper_29953101922703
// MI455X (gfx1250) — matrix operand order repaired and checked by reading, not compiled, not run on hardware
//
#include <hip/hip_runtime.h>
#include <hip/hip_bf16.h>
#include <math.h>

// ---------------------------------------------------------------------------
// SchNet forward on MI455X (gfx1250), wave32 + WMMA f32_16x16x32_f16.
// GEMMs via v_wmma; weight/feature panels staged into LDS with
// global_load_async_to_lds_b128 (ASYNCcnt); big-LDS (~112KB) pair kernel.
// ---------------------------------------------------------------------------

typedef _Float16 h16;
typedef __attribute__((ext_vector_type(8)))  _Float16 v8h;
typedef __attribute__((ext_vector_type(16))) _Float16 v16h;
typedef __attribute__((ext_vector_type(8)))  float    v8f;

#define NATOMS_TOTAL 32768
#define NMOL 1024
#define APM  32        // atoms per molecule
#define D    128
#define NRBF 50
#define KPAD 64        // Wf1 K padded 50 -> 64
#define NLAYER 6
#define RBF_COEFF (-0.5f * (49.0f/5.0f) * (49.0f/5.0f))
#define PI_F 3.14159265358979f

union AFrag { v16h v; struct { v8h lo, hi; } p; h16 e[16]; };

__device__ __forceinline__ void wmma_f16(v16h a, v16h b, v8f& c) {
  c = __builtin_amdgcn_wmma_f32_16x16x32_f16(false, a, false, b, (short)0, c,
                                             false, false);
}

// Branchless shifted softplus: softplus(v) - ln2, with
// softplus(v) = max(v,0) + log1p(exp(-|v|)).  exp(-|v|) <= 1 -> no overflow,
// no guard, no divergent branch; pure TRANS ops co-execute with XDL WMMA.
__device__ __forceinline__ float ssp_f(float v) {
  float sp = fmaxf(v, 0.f) + __logf(1.f + __expf(-fabsf(v)));
  return sp - 0.69314718056f;
}

// ---- CDNA5 async LDS<->global DMA (ASYNCcnt) ------------------------------
// Generic shared pointers carry the LDS byte offset in their low 32 bits
// (aperture scheme, ISA 10.2), so truncation yields the DS address VGPR.
__device__ __forceinline__ void async_ld_b128(void* lds, const void* g) {
  asm volatile("global_load_async_to_lds_b128 %0, %1, off"
               :: "v"((unsigned)(uintptr_t)lds), "v"(g) : "memory");
}
__device__ __forceinline__ void async_st_b128(void* g, const void* lds) {
  asm volatile("global_store_async_from_lds_b128 %0, %1, off"
               :: "v"(g), "v"((unsigned)(uintptr_t)lds) : "memory");
}
__device__ __forceinline__ void wait_async0() {
  asm volatile("s_wait_asynccnt 0" ::: "memory");
}

// A fragment (16x32, f16) from a row-major f16 matrix with 128-col stride.
// Lane m=l&15 holds row m; lo half = K [kb,kb+8), hi half = K [kb+16,kb+24),
// kb = (l>>4)*8 + ks*32.   (ISA 7.12.2)
__device__ __forceinline__ v16h load_afrag_rowmajor(const h16* __restrict__ A,
                                                    int row, int kb) {
  AFrag f;
  f.p.lo = *(const v8h*)(A + row * D + kb);
  f.p.hi = *(const v8h*)(A + row * D + kb + 16);
  return f.v;
}

// B fragment (32x16, f16) from an n-major panel: Bt[n][k] = B[k][n];
// lane n=l&15 holds col n; lo half = K [kb,kb+8), hi half = K [kb+16,kb+24),
// kb = (l>>4)*8 + ks*32.
__device__ __forceinline__ v16h load_bfrag_nmajor(const h16* __restrict__ Bt,
                                                  int col, int kb, int kstride) {
  AFrag f;
  f.p.lo = *(const v8h*)(Bt + col * kstride + kb);
  f.p.hi = *(const v8h*)(Bt + col * kstride + kb + 16);
  return f.v;
}

// ---------------------------------------------------------------------------
// Prep kernels
// ---------------------------------------------------------------------------

// W: [L][128][128] (k-major) f32 -> Wt: [L][128][128] (n-major) f16
__global__ void transpose_w_kernel(const float* __restrict__ W,
                                   h16* __restrict__ Wt, int total) {
  int idx = blockIdx.x * blockDim.x + threadIdx.x;
  if (idx >= total) return;
  int l = idx >> 14, rem = idx & 16383, n = rem >> 7, k = rem & 127;
  Wt[idx] = (h16)W[l * 16384 + k * 128 + n];
}

// Wf1: [6][50][128] f32 -> Wp: [6][128][64] f16, K zero-padded 50->64
__global__ void pad_wf1_kernel(const float* __restrict__ W,
                               h16* __restrict__ Wp) {
  int idx = blockIdx.x * blockDim.x + threadIdx.x;
  if (idx >= NLAYER * D * KPAD) return;
  int l = idx / (D * KPAD), rem = idx % (D * KPAD), n = rem >> 6, k = rem & 63;
  Wp[idx] = (h16)((k < NRBF) ? W[l * NRBF * D + k * D + n] : 0.f);
}

// x[n][c] = emb[z[n]][c], plus f16 copy
__global__ void embed_kernel(const int* __restrict__ z,
                             const float* __restrict__ emb,
                             float* __restrict__ x, h16* __restrict__ xh) {
  int idx = blockIdx.x * blockDim.x + threadIdx.x;
  if (idx >= NATOMS_TOTAL * D) return;
  int n = idx >> 7, c = idx & 127;
  float v = emb[z[n] * D + c];
  x[idx] = v;
  xh[idx] = (h16)v;
}

// ---------------------------------------------------------------------------
// Generic [M,128] x [128,128] WMMA GEMM. B panel (32KB f16) is DMA'd into
// LDS once per block via async loads; one wave = 16 rows x 128 cols
// (8 n-tiles x 4 k-steps = 32 v_wmma per wave, A frags reused 8x).
// MODE 0: out_f32 = A@B                      (in2f -> xf)
// MODE 1: out_f16 = ssp(A@B + bias)          (f2out stage 1 -> t)
// MODE 2: resid  += A@B + bias; out_f16=f16(resid)   (f2out stage 2 -> x,xh)
// ---------------------------------------------------------------------------
template <int MODE>
__global__ __launch_bounds__(256)
__attribute__((amdgpu_waves_per_eu(4))) void gemm128_kernel(
    const h16* __restrict__ A, const h16* __restrict__ Bt,
    const float* __restrict__ bias, float* __restrict__ outF,
    h16* __restrict__ outH, float* __restrict__ resid) {
  __shared__ __align__(32) h16 bt_s[D * D];  // 32 KB

  // async DMA the whole B panel into LDS (2048 x 16B chunks)
  for (int c = threadIdx.x; c < 2048; c += 256)
    async_ld_b128((char*)bt_s + c * 16, (const char*)Bt + c * 16);
  wait_async0();
  __syncthreads();

  int wave = (blockIdx.x * blockDim.x + threadIdx.x) >> 5;
  int lane = threadIdx.x & 31;
  int half = lane >> 4, ln = lane & 15;
  int row0 = wave * 16;
  int mrow = row0 + ln;

  __builtin_prefetch(A + (mrow + 128) * D, 0, 1);  // global_prefetch_b8

  // hoist per-lane bias values out of the tile loop
  float bv[8];
#pragma unroll
  for (int nt = 0; nt < 8; ++nt) bv[nt] = (MODE == 0) ? 0.f : bias[nt * 16 + ln];

  v16h a[4];
#pragma unroll
  for (int ks = 0; ks < 4; ++ks)
    a[ks] = load_afrag_rowmajor(A, mrow, half * 8 + ks * 32);

#pragma unroll 4
  for (int nt = 0; nt < 8; ++nt) {
    v8f c = {};
#pragma unroll
    for (int ks = 0; ks < 4; ++ks) {
      v16h b = load_bfrag_nmajor(bt_s, nt * 16 + ln, half * 8 + ks * 32, D);
      wmma_f16(a[ks], b, c);
    }
    int col = nt * 16 + ln;
#pragma unroll
    for (int v = 0; v < 8; ++v) {
      int r = row0 + v + half * 8;
      float val = c[v] + bv[nt];
      if (MODE == 0) {
        outF[r * D + col] = val;
      } else if (MODE == 1) {
        outH[r * D + col] = (h16)ssp_f(val);
      } else {
        float xn = resid[r * D + col] + val;
        resid[r * D + col] = xn;
        outH[r * D + col] = (h16)xn;
      }
    }
  }
}

// ---------------------------------------------------------------------------
// Pair kernel: one workgroup (4 waves) per molecule. ~112KB LDS (gfx1250 WGP
// has 320KB). Weights + xf staged by async DMA; d/rcut precomputed; RBF
// A-frags synthesized in registers; GEMM1 -> ssp -> LDS staging -> GEMM2 ->
// rcut -> message*xf -> segment-sum via ds_add_f32; agg written back with
// async store-from-LDS.
// ---------------------------------------------------------------------------
__global__ __launch_bounds__(128)
__attribute__((amdgpu_waves_per_eu(4))) void pair_kernel(
    const float* __restrict__ pos, const float* __restrict__ xf,
    const h16* __restrict__ w1p,   // [128][64]  n-major, K padded
    const h16* __restrict__ w2t,   // [128][128] n-major
    const float* __restrict__ bf1, const float* __restrict__ bf2,
    h16* __restrict__ aggh) {
  __shared__ __align__(16) float pos_s[APM * 3];
  __shared__ __align__(16) float d_s[APM * APM];
  __shared__ __align__(16) float rc_s[APM * APM];
  __shared__ __align__(16) float xf_s[APM * D];     // 16 KB
  __shared__ __align__(16) float agg_s[APM * D];    // 16 KB
  __shared__ __align__(16) h16 aggh_s[APM * D];     //  8 KB
  __shared__ __align__(32) h16 w1_s[D * KPAD];      // 16 KB
  __shared__ __align__(32) h16 w2_s[D * D];         // 32 KB
  __shared__ __align__(32) h16 h1_s[4][16 * D];     // 16 KB

  int mol = blockIdx.x;
  int tid = threadIdx.x;
  int lane = tid & 31, wv = tid >> 5;
  int half = lane >> 4, ln = lane & 15;

  // --- async DMA: weight panels + this molecule's xf into LDS ---
  for (int c = tid; c < 1024; c += 128)   // w1: 16KB
    async_ld_b128((char*)w1_s + c * 16, (const char*)w1p + c * 16);
  for (int c = tid; c < 2048; c += 128)   // w2: 32KB
    async_ld_b128((char*)w2_s + c * 16, (const char*)w2t + c * 16);
  {
    const char* xfg = (const char*)(xf + (size_t)mol * APM * D);
    for (int c = tid; c < 1024; c += 128)  // xf: 16KB
      async_ld_b128((char*)xf_s + c * 16, xfg + c * 16);
  }

  if (tid < APM * 3) pos_s[tid] = pos[mol * APM * 3 + tid];
  for (int i = tid; i < APM * D; i += 128) agg_s[i] = 0.f;

  // hoist per-lane bias values (constant across all M-tiles)
  float b1v[8], b2v[8];
#pragma unroll
  for (int nt = 0; nt < 8; ++nt) {
    b1v[nt] = bf1[nt * 16 + ln];
    b2v[nt] = bf2[nt * 16 + ln];
  }

  wait_async0();
  __syncthreads();

  for (int p = tid; p < APM * APM; p += 128) {
    int i = p >> 5, j = p & 31;
    float dx = pos_s[j * 3 + 0] - pos_s[i * 3 + 0];
    float dy = pos_s[j * 3 + 1] - pos_s[i * 3 + 1];
    float dz = pos_s[j * 3 + 2] - pos_s[i * 3 + 2];
    float d = sqrtf(dx * dx + dy * dy + dz * dz);
    d_s[p] = d;
    float rc = 0.5f * (__cosf(d * (PI_F / 5.0f)) + 1.f);
    rc_s[p] = (i != j && d < 5.0f) ? rc : 0.f;
  }
  __syncthreads();

  // 64 M-tiles of 16 pairs; uniform per-wave control flow -> EXEC stays full.
#pragma unroll 1
  for (int mt = wv; mt < 64; mt += 4) {
    int p0 = mt * 16;
    int irow = p0 >> 5;       // receiver atom i (constant over tile)
    int jbase = p0 & 31;      // j = jbase + m
    float dm = d_s[p0 + ln];  // distance for this lane's A-row

    // --- GEMM1 A fragments: Gaussian RBF, synthesized in registers ---
    v16h a1[2];
#pragma unroll
    for (int ks = 0; ks < 2; ++ks) {
      AFrag f;
#pragma unroll
      for (int idx = 0; idx < 16; ++idx) {
        int koff = (idx < 8) ? idx : idx + 8;  // lo/hi half K offsets
        int k = ks * 32 + half * 8 + koff;
        float t = dm - (float)k * (5.0f / 49.0f);
        float r = __expf(RBF_COEFF * t * t);   // always computed
        r = (k < NRBF) ? r : 0.f;              // cheap v_cndmask, no branch
        f.e[idx] = (h16)r;
      }
      a1[ks] = f.v;
    }

    // --- GEMM1 [16,64]x[64,128] + bias + ssp -> per-wave LDS (f16) ---
#pragma unroll 2
    for (int nt = 0; nt < 8; ++nt) {
      v8f c = {};
#pragma unroll
      for (int ks = 0; ks < 2; ++ks) {
        v16h b = load_bfrag_nmajor(w1_s, nt * 16 + ln, half * 8 + ks * 32, KPAD);
        wmma_f16(a1[ks], b, c);
      }
      int col = nt * 16 + ln;
#pragma unroll
      for (int v = 0; v < 8; ++v) {
        int m = v + half * 8;  // D layout -> row m, col col
        h1_s[wv][m * D + col] = (h16)ssp_f(c[v] + b1v[nt]);
      }
    }
    // h1_s[wv] is wave-private; DS ops are in-order within a wave.

    // --- GEMM2 A fragments from staged H1 (D-layout -> A-layout) ---
    v16h a2[4];
#pragma unroll
    for (int ks = 0; ks < 4; ++ks) {
      AFrag f;
      const h16* base = &h1_s[wv][ln * D + half * 8 + ks * 32];
      f.p.lo = *(const v8h*)(base);
      f.p.hi = *(const v8h*)(base + 16);
      a2[ks] = f.v;
    }

    float rcv[8];
#pragma unroll
    for (int v = 0; v < 8; ++v) rcv[v] = rc_s[p0 + v + half * 8];

    // --- GEMM2 [16,128]x[128,128] + bias, *rcut, message, segment-sum ---
#pragma unroll 2
    for (int nt = 0; nt < 8; ++nt) {
      v8f c = {};
#pragma unroll
      for (int ks = 0; ks < 4; ++ks) {
        v16h b = load_bfrag_nmajor(w2_s, nt * 16 + ln, half * 8 + ks * 32, D);
        wmma_f16(a2[ks], b, c);
      }
      int col = nt * 16 + ln;
      float partial = 0.f;
#pragma unroll
      for (int v = 0; v < 8; ++v) {
        int m = v + half * 8;
        float W = (c[v] + b2v[nt]) * rcv[v];
        partial += W * xf_s[(jbase + m) * D + col];
      }
      atomicAdd(&agg_s[irow * D + col], partial);  // ds_add_f32
    }
  }
  __syncthreads();

  // convert to f16 in LDS, then async store-from-LDS back to global
  for (int i = tid; i < APM * D; i += 128) aggh_s[i] = (h16)agg_s[i];
  __syncthreads();
  {
    char* aggg = (char*)(aggh + (size_t)mol * APM * D);
    for (int c = tid; c < 512; c += 128)  // 8KB
      async_st_b128(aggg + c * 16, (const char*)aggh_s + c * 16);
  }
  wait_async0();
}

// ---------------------------------------------------------------------------
// Head: mol = sum_atoms x; pred = silu(mol@Wh1+bh1)@Wh2 + bh2.  (~38 MFLOP)
// ---------------------------------------------------------------------------
__global__ __launch_bounds__(128) void head_kernel(
    const float* __restrict__ x, const float* __restrict__ Wh1,
    const float* __restrict__ bh1, const float* __restrict__ Wh2,
    const float* __restrict__ bh2, float* __restrict__ out) {
  __shared__ float mol_s[D];
  __shared__ float h_s[D];
  int mol = blockIdx.x, c = threadIdx.x;
  float s = 0.f;
  for (int a = 0; a < APM; ++a) s += x[(mol * APM + a) * D + c];
  mol_s[c] = s;
  __syncthreads();
  float acc = bh1[c];
  for (int k = 0; k < D; ++k) acc += mol_s[k] * Wh1[k * D + c];
  float hv = acc / (1.f + __expf(-acc));  // silu
  h_s[c] = hv * Wh2[c];
  __syncthreads();
  if (c == 0) {
    float p = bh2[0];
    for (int k = 0; k < D; ++k) p += h_s[k];
    out[mol] = p;
  }
}

// ---------------------------------------------------------------------------
// Host driver
// ---------------------------------------------------------------------------
extern "C" void kernel_launch(void* const* d_in, const int* in_sizes, int n_in,
                              void* d_out, int out_size, void* d_ws,
                              size_t ws_size, hipStream_t stream) {
  (void)in_sizes; (void)n_in; (void)out_size; (void)ws_size;
  const int*   z    = (const int*)d_in[0];
  const float* pos  = (const float*)d_in[1];
  const float* emb  = (const float*)d_in[6];
  const float* Wi   = (const float*)d_in[7];
  const float* Wf1  = (const float*)d_in[8];
  const float* bf1  = (const float*)d_in[9];
  const float* Wf2  = (const float*)d_in[10];
  const float* bf2  = (const float*)d_in[11];
  const float* Wo1  = (const float*)d_in[12];
  const float* bo1  = (const float*)d_in[13];
  const float* Wo2  = (const float*)d_in[14];
  const float* bo2  = (const float*)d_in[15];
  const float* Wh1  = (const float*)d_in[16];
  const float* bh1  = (const float*)d_in[17];
  const float* Wh2  = (const float*)d_in[18];
  const float* bh2  = (const float*)d_in[19];
  float* pred = (float*)d_out;

  // workspace carve-up
  char* ws = (char*)d_ws;
  size_t o = 0;
  float* x    = (float*)(ws + o); o += (size_t)NATOMS_TOTAL * D * 4;   // 16 MB
  h16*   xh   = (h16*)  (ws + o); o += (size_t)NATOMS_TOTAL * D * 2;   //  8 MB
  float* xf   = (float*)(ws + o); o += (size_t)NATOMS_TOTAL * D * 4;   // 16 MB
  h16*   aggh = (h16*)  (ws + o); o += (size_t)NATOMS_TOTAL * D * 2;   //  8 MB
  h16*   th   = (h16*)  (ws + o); o += (size_t)NATOMS_TOTAL * D * 2;   //  8 MB
  h16*   WiT  = (h16*)  (ws + o); o += (size_t)NLAYER * D * D * 2;
  h16*   Wf2T = (h16*)  (ws + o); o += (size_t)NLAYER * D * D * 2;
  h16*   Wo1T = (h16*)  (ws + o); o += (size_t)NLAYER * D * D * 2;
  h16*   Wo2T = (h16*)  (ws + o); o += (size_t)NLAYER * D * D * 2;
  h16*   Wf1P = (h16*)  (ws + o); o += (size_t)NLAYER * D * KPAD * 2;

  // prep: weight conversion/transpose + embedding
  int wTot = NLAYER * D * D;
  transpose_w_kernel<<<(wTot + 255) / 256, 256, 0, stream>>>(Wi, WiT, wTot);
  transpose_w_kernel<<<(wTot + 255) / 256, 256, 0, stream>>>(Wf2, Wf2T, wTot);
  transpose_w_kernel<<<(wTot + 255) / 256, 256, 0, stream>>>(Wo1, Wo1T, wTot);
  transpose_w_kernel<<<(wTot + 255) / 256, 256, 0, stream>>>(Wo2, Wo2T, wTot);
  pad_wf1_kernel<<<(NLAYER * D * KPAD + 255) / 256, 256, 0, stream>>>(Wf1, Wf1P);
  embed_kernel<<<(NATOMS_TOTAL * D + 255) / 256, 256, 0, stream>>>(z, emb, x, xh);

  // 2048 M-tiles of 16 rows, 8 waves per 256-thread block -> 256 blocks
  const int gemmBlocks = (NATOMS_TOTAL / 16) / 8;
  for (int l = 0; l < NLAYER; ++l) {
    gemm128_kernel<0><<<gemmBlocks, 256, 0, stream>>>(
        xh, WiT + (size_t)l * D * D, nullptr, xf, nullptr, nullptr);
    pair_kernel<<<NMOL, 128, 0, stream>>>(
        pos, xf, Wf1P + (size_t)l * D * KPAD, Wf2T + (size_t)l * D * D,
        bf1 + l * D, bf2 + l * D, aggh);
    gemm128_kernel<1><<<gemmBlocks, 256, 0, stream>>>(
        aggh, Wo1T + (size_t)l * D * D, bo1 + l * D, nullptr, th, nullptr);
    gemm128_kernel<2><<<gemmBlocks, 256, 0, stream>>>(
        th, Wo2T + (size_t)l * D * D, bo2 + l * D, nullptr, xh, x);
  }
  head_kernel<<<NMOL, 128, 0, stream>>>(x, Wh1, bh1, Wh2, bh2, pred);
}
